// MPNN_84473416778294
// MI455X (gfx1250) — matrix operand order repaired and checked by reading, not compiled, not run on hardware
//
#include <hip/hip_runtime.h>
#include <hip/hip_bf16.h>
#include <stdint.h>

typedef __attribute__((ext_vector_type(16))) _Float16 v16h;
typedef __attribute__((ext_vector_type(8)))  _Float16 v8h;
typedef __attribute__((ext_vector_type(8)))  float    v8f;
typedef __attribute__((ext_vector_type(4)))  unsigned int v4u;
typedef __attribute__((ext_vector_type(8)))  int v8i;
typedef __attribute__((ext_vector_type(4)))  int v4i;

#define NN 10000
#define EE 50000
#define NODE_DIM 32
#define EDGE_DIM 16
#define HH 64
#define LL 3
#define GG 256
#define EPSV 1e-5f

#if defined(__has_builtin)
#if __has_builtin(__builtin_amdgcn_tensor_load_to_lds)
#define HAS_TDM 1
#endif
#endif
#ifndef HAS_TDM
#define HAS_TDM 0
#endif

// ---------------------------------------------------------------- utilities
__global__ void k_zero(float* p, int n) {
  int i = blockIdx.x * blockDim.x + threadIdx.x;
  for (; i < n; i += gridDim.x * blockDim.x) p[i] = 0.0f;
}

__global__ void k_deg(const int* __restrict__ ei, float* __restrict__ deg) {
  int e = blockIdx.x * blockDim.x + threadIdx.x;
  if (e < EE) atomicAdd(&deg[ei[EE + e]], 1.0f);
}

__global__ void k_gcnt(const int* __restrict__ batch, float* __restrict__ gcnt) {
  int n = blockIdx.x * blockDim.x + threadIdx.x;
  if (n < NN) atomicAdd(&gcnt[batch[n]], 1.0f);
}

// ------------------------------------------------- node embedding: h = relu(x@W+b)
__global__ __launch_bounds__(256) void k_node_emb(
    const float* __restrict__ x, const float* __restrict__ w,
    const float* __restrict__ b, float* __restrict__ h) {
  int t = threadIdx.x;
  int o = t & 63;
  int n = blockIdx.x * 4 + (t >> 6);
  float acc = b[o];
  const float* xr = x + (size_t)n * NODE_DIM;
#pragma unroll
  for (int d = 0; d < NODE_DIM; ++d) acc += xr[d] * w[d * HH + o];
  h[(size_t)n * HH + o] = acc > 0.f ? acc : 0.f;
}

// ---------------------- convert edge_w2 (+ b2 as slice 64) into f16 [65][o][h]
__global__ void k_convert_w2(const float* __restrict__ w2,
                             const float* __restrict__ b2,
                             _Float16* __restrict__ w2t) {
  int idx = blockIdx.x * blockDim.x + threadIdx.x;  // over LL*65*4096
  int l = idx / (65 * 4096);
  int r = idx - l * 65 * 4096;
  int k = r >> 12;
  int j = r & 4095;
  int o = j >> 6, hh = j & 63;
  float v = (k < 64) ? w2[((size_t)(l * 64 + k) << 12) + hh * 64 + o]
                     : b2[(size_t)l * 4096 + hh * 64 + o];
  w2t[(((size_t)(l * 65 + k) * 64 + o) << 6) + hh] = (_Float16)v;
}

// ---------------- edge MLP layer-1 (a = relu(ea@w1+b1)) + gather h[src], both f16
__global__ __launch_bounds__(256) void k_edge_mlp1(
    const float* __restrict__ ea, const float* __restrict__ w1,
    const float* __restrict__ b1, const float* __restrict__ h,
    const int* __restrict__ ei, _Float16* __restrict__ a16,
    _Float16* __restrict__ hs16) {
  int t = threadIdx.x;
  int k = t & 63;
  int e = blockIdx.x * 4 + (t >> 6);
  float acc = b1[k];
  const float* er = ea + (size_t)e * EDGE_DIM;
#pragma unroll
  for (int d = 0; d < EDGE_DIM; ++d) acc += er[d] * w1[d * HH + k];
  a16[(size_t)e * HH + k] = (_Float16)(acc > 0.f ? acc : 0.f);
  int s = ei[e];
  hs16[(size_t)e * HH + k] = (_Float16)h[(size_t)s * HH + k];
}

// ---------------------------------------------------------------- WMMA GEMM core
__device__ inline v16h ld_fragA(const _Float16* row, int base) {
  // A(16x32) lane layout: halfs [base..base+7] then [base+16..base+23]
  v8h lo = *(const v8h*)(row + base);
  v8h hi = *(const v8h*)(row + base + 16);
  v16h r;
#pragma unroll
  for (int i = 0; i < 8; ++i) { r[i] = lo[i]; r[8 + i] = hi[i]; }
  return r;
}
__device__ inline v16h ld_fragB(const _Float16* p) {
  // B(32x16) lane layout: halfs [p..p+7] then [p+16..p+23]
  v8h lo = *(const v8h*)p;
  v8h hi = *(const v8h*)(p + 16);
  v16h r;
#pragma unroll
  for (int i = 0; i < 8; ++i) { r[i] = lo[i]; r[8 + i] = hi[i]; }
  return r;
}

#define WAVES 8
#define KB 4
#define BSTRIDE 72  // padded LDS row stride (halfs): 128B row + 16B pad
#define NT (EE / 16)                 // 3125 16-edge tiles
#define W2BUF (KB * 64 * BSTRIDE)    // halfs per staging buffer
#define SMEM_HALFS (2 * W2BUF + WAVES * 32 * 64)
#define SMEM_BYTES (SMEM_HALFS * 2)  // 106,496 B dynamic LDS (WGP has 320KB)

#if HAS_TDM
// Tensor Data Mover: 2D tile, rows of 32 DWORDs (64 halfs), LDS padding of
// 4 DWORDs every 32 DWORDs -> reproduces the 144B padded row stride.
__device__ inline void tdm_load_w2(const _Float16* gsrc, unsigned lds_addr,
                                   int rows) {
  unsigned long long ga = (unsigned long long)(uintptr_t)gsrc;
  v4u g0 = {1u,                                   // count=1, is_restore=0
            lds_addr,                             // [63:32] lds_addr
            (unsigned)ga,                         // global_addr lo
            (unsigned)((ga >> 32) & 0x01ffffffu) | (2u << 30)};  // type=2
  v8i g1 = {(2 << 16) | (1 << 20) | (4 << 22) | (3 << 25),
            // data_size=4B, pad_enable, pad_interval=32dw, pad_amount=4dw
            (int)(32u << 16),        // tensor_dim0 = 32 dwords
            (int)(16384u << 16),     // tensor_dim1 (large; no OOB clamp)
            (int)(32u << 16),        // tile_dim0 = 32 dwords
            rows,                    // tile_dim1 = rows
            32,                      // tensor_dim0_stride = 32 dwords
            0, 0};
  v4i z4 = {0, 0, 0, 0};
#if defined(__clang_major__) && __clang_major__ >= 23
  v8i z8 = {0, 0, 0, 0, 0, 0, 0, 0};
  __builtin_amdgcn_tensor_load_to_lds(g0, g1, z4, z4, z8, 0);
#else
  __builtin_amdgcn_tensor_load_to_lds(g0, g1, z4, z4, 0);
#endif
}
#endif

__global__ __launch_bounds__(256) void k_msg_gemm(
    const _Float16* __restrict__ a16, const _Float16* __restrict__ hs16,
    const _Float16* __restrict__ w2t /* [65][64][64] this layer */,
    const int* __restrict__ ei, float* __restrict__ aggr) {
  extern __shared__ _Float16 smem[];
  _Float16* w2blk = smem;                 // [2][W2BUF]
  _Float16* atile = smem + 2 * W2BUF;     // [WAVES][32][64]

  const int tid = threadIdx.x;
  const int wave = tid >> 5, lane = tid & 31;
  const int m = lane & 15, hi = lane >> 4;
  const int pairId = blockIdx.x * WAVES + wave;
  const int t0 = pairId * 2;              // two 16-edge tiles per wave
  const bool act0 = t0 < NT;
  const bool act1 = (t0 + 1) < NT;
  const int t0c = act0 ? t0 : (NT - 1);   // clamp for loads (stores guarded)
  const int t1c = act1 ? (t0 + 1) : (NT - 1);
  const int e0 = t0c * 16, e1 = t1c * 16;

  v8f c0[4] = {}, c1[4] = {};

  // stage both a-tiles [16][64] into LDS (clamped -> always safe)
  {
    _Float16* at = atile + wave * 32 * 64;
    const _Float16* ar0 = a16 + (size_t)e0 * HH;
    const _Float16* ar1 = a16 + (size_t)e1 * HH;
    for (int i = lane; i < 128; i += 32) {
      *(v8h*)(at + i * 8) = *(const v8h*)(ar0 + i * 8);
      *(v8h*)(at + 1024 + i * 8) = *(const v8h*)(ar1 + i * 8);
    }
  }
  // hs fragments (constant across k) straight from global
  const _Float16* hr0 = hs16 + ((size_t)e0 + m) * HH;
  const _Float16* hr1 = hs16 + ((size_t)e1 + m) * HH;
  v16h hf00 = ld_fragA(hr0, hi * 8);
  v16h hf01 = ld_fragA(hr0, 32 + hi * 8);
  v16h hf10 = ld_fragA(hr1, hi * 8);
  v16h hf11 = ld_fragA(hr1, 32 + hi * 8);

#if HAS_TDM
  if (wave == 0)
    tdm_load_w2(w2t, (unsigned)(uintptr_t)w2blk, KB * 64);
#endif

  const int nblk = (65 + KB - 1) / KB;
  for (int ib = 0; ib < nblk; ++ib) {
    const int kb = ib * KB;
    const int nk = (65 - kb) < KB ? (65 - kb) : KB;
    _Float16* bufp = w2blk + (ib & 1) * W2BUF;

#if HAS_TDM
    if (wave == 0) __builtin_amdgcn_s_wait_tensorcnt(0);
    __syncthreads();  // buffer ready; previous compute on other buf done
    {
      const int kb2 = kb + KB;
      if (wave == 0 && kb2 < 65) {
        const int nk2 = (65 - kb2) < KB ? (65 - kb2) : KB;
        tdm_load_w2(w2t + ((size_t)kb2 << 12),
                    (unsigned)(uintptr_t)(w2blk + ((ib & 1) ^ 1) * W2BUF),
                    nk2 * 64);
      }
    }
#else
    __syncthreads();
    for (int i = tid; i < nk * 64 * 8; i += 256) {
      int slice = i >> 9;
      int rem = i & 511;
      int o = rem >> 3, ch = rem & 7;
      *(v8h*)(bufp + (slice * 64 + o) * BSTRIDE + ch * 8) =
          *(const v8h*)(w2t + (((size_t)(kb + slice) * 64 + o) << 6) + ch * 8);
    }
    __syncthreads();
#endif

    const _Float16* at0 = atile + (wave * 32 + m) * 64;
    const _Float16* at1 = at0 + 16 * 64;
    for (int kk = 0; kk < nk; ++kk) {
      _Float16 a0 = at0[kb + kk];  // LDS broadcast a[m, k], tile 0
      _Float16 a1 = at1[kb + kk];  // tile 1
      v16h A00, A01, A10, A11;
#pragma unroll
      for (int i = 0; i < 16; ++i) {
        A00[i] = hf00[i] * a0; A01[i] = hf01[i] * a0;
        A10[i] = hf10[i] * a1; A11[i] = hf11[i] * a1;
      }
      const _Float16* bbase = bufp + kk * 64 * BSTRIDE;
#pragma unroll
      for (int t4 = 0; t4 < 4; ++t4) {
        const _Float16* bp = bbase + (t4 * 16 + m) * BSTRIDE + hi * 8;
        v16h B0 = ld_fragB(bp);       // shared by both M-tiles
        v16h B1 = ld_fragB(bp + 32);
        c0[t4] = __builtin_amdgcn_wmma_f32_16x16x32_f16(false, A00, false, B0,
                                                        (short)0, c0[t4], false, false);
        c0[t4] = __builtin_amdgcn_wmma_f32_16x16x32_f16(false, A01, false, B1,
                                                        (short)0, c0[t4], false, false);
        c1[t4] = __builtin_amdgcn_wmma_f32_16x16x32_f16(false, A10, false, B0,
                                                        (short)0, c1[t4], false, false);
        c1[t4] = __builtin_amdgcn_wmma_f32_16x16x32_f16(false, A11, false, B1,
                                                        (short)0, c1[t4], false, false);
      }
    }
    __syncthreads();  // compute on bufp finished before it is refilled
  }

  const int* dstIdx = ei + EE;
  if (act0) {
#pragma unroll
    for (int t4 = 0; t4 < 4; ++t4) {
      int o = t4 * 16 + m;
#pragma unroll
      for (int v = 0; v < 8; ++v) {
        int e = e0 + hi * 8 + v;  // C layout: M = v + 8*hi
        atomicAdd(aggr + (size_t)dstIdx[e] * HH + o, c0[t4][v]);
      }
    }
  }
  if (act1) {
#pragma unroll
    for (int t4 = 0; t4 < 4; ++t4) {
      int o = t4 * 16 + m;
#pragma unroll
      for (int v = 0; v < 8; ++v) {
        int e = e1 + hi * 8 + v;
        atomicAdd(aggr + (size_t)dstIdx[e] * HH + o, c1[t4][v]);
      }
    }
  }
}

// --------- hnew = aggr/deg + h@root_w + conv_b ; accumulate BN sum/sumsq
__global__ __launch_bounds__(256) void k_root(
    const float* __restrict__ aggr, const float* __restrict__ deg,
    const float* __restrict__ h, const float* __restrict__ rootw,
    const float* __restrict__ convb, float* __restrict__ hnew,
    float* __restrict__ bns /* [128] */) {
  __shared__ float red[512];
  int t = threadIdx.x;
  int o = t & 63, sub = t >> 6;
  float lsum = 0.f, lsq = 0.f;
  for (int i = 0; i < 10; ++i) {
    int n = blockIdx.x * 40 + i * 4 + sub;  // 250 blocks * 40 = 10000 exact
    float acc = convb[o];
    const float* hr = h + (size_t)n * HH;
#pragma unroll
    for (int k = 0; k < HH; ++k) acc += hr[k] * rootw[k * HH + o];
    float d = deg[n]; if (d < 1.f) d = 1.f;
    acc += aggr[(size_t)n * HH + o] / d;
    hnew[(size_t)n * HH + o] = acc;
    lsum += acc; lsq += acc * acc;
  }
  red[t] = lsum; red[256 + t] = lsq;
  __syncthreads();
  if (t < 64) {
    float s = red[t] + red[t + 64] + red[t + 128] + red[t + 192];
    float q = red[256 + t] + red[320 + t] + red[384 + t] + red[448 + t];
    atomicAdd(&bns[o], s);
    atomicAdd(&bns[64 + o], q);
  }
}

// --------- batchnorm (population stats) + relu, write back into h
__global__ __launch_bounds__(256) void k_bn_relu(
    const float* __restrict__ hnew, const float* __restrict__ bns,
    const float* __restrict__ g, const float* __restrict__ b,
    float* __restrict__ h) {
  int t = threadIdx.x;
  int o = t & 63;
  int n = blockIdx.x * 4 + (t >> 6);
  float mu = bns[o] * (1.0f / NN);
  float var = bns[64 + o] * (1.0f / NN) - mu * mu;
  float v = g[o] * (hnew[(size_t)n * HH + o] - mu) * rsqrtf(var + EPSV) + b[o];
  h[(size_t)n * HH + o] = v > 0.f ? v : 0.f;
}

// --------- global mean pool accumulate
__global__ __launch_bounds__(256) void k_pool(
    const float* __restrict__ h, const int* __restrict__ batch,
    float* __restrict__ pooled) {
  int t = threadIdx.x;
  int o = t & 63;
  int n = blockIdx.x * 4 + (t >> 6);
  atomicAdd(&pooled[(size_t)batch[n] * HH + o], h[(size_t)n * HH + o]);
}

// --------- readout MLP per graph
__global__ __launch_bounds__(64) void k_readout(
    const float* __restrict__ pooled, const float* __restrict__ gcnt,
    const float* __restrict__ w1, const float* __restrict__ b1,
    const float* __restrict__ w2, const float* __restrict__ b2,
    float* __restrict__ out) {
  __shared__ float pl[64];
  __shared__ float tt[32];
  int g = blockIdx.x, t = threadIdx.x;
  float c = gcnt[g]; if (c < 1.f) c = 1.f;
  pl[t] = pooled[(size_t)g * HH + t] / c;
  __syncthreads();
  if (t < 32) {
    float acc = b1[t];
#pragma unroll
    for (int o = 0; o < 64; ++o) acc += pl[o] * w1[o * 32 + t];
    tt[t] = acc > 0.f ? acc : 0.f;
  }
  __syncthreads();
  if (t == 0) {
    float acc = b2[0];
#pragma unroll
    for (int j = 0; j < 32; ++j) acc += tt[j] * w2[j];
    out[g] = acc;
  }
}

// ================================================================ launch
extern "C" void kernel_launch(void* const* d_in, const int* in_sizes, int n_in,
                              void* d_out, int out_size, void* d_ws, size_t ws_size,
                              hipStream_t stream) {
  const float* x        = (const float*)d_in[0];
  const int*   ei       = (const int*)d_in[1];
  const float* eattr    = (const float*)d_in[2];
  const int*   batch    = (const int*)d_in[3];
  const float* nemb_w   = (const float*)d_in[4];
  const float* nemb_b   = (const float*)d_in[5];
  const float* ew1      = (const float*)d_in[6];
  const float* eb1      = (const float*)d_in[7];
  const float* ew2      = (const float*)d_in[8];
  const float* eb2      = (const float*)d_in[9];
  const float* rootw    = (const float*)d_in[10];
  const float* convb    = (const float*)d_in[11];
  const float* bng      = (const float*)d_in[12];
  const float* bnb      = (const float*)d_in[13];
  const float* row1     = (const float*)d_in[14];
  const float* rob1     = (const float*)d_in[15];
  const float* row2     = (const float*)d_in[16];
  const float* rob2     = (const float*)d_in[17];
  float* out = (float*)d_out;

  size_t off = 0;
  auto alloc = [&](size_t bytes) -> void* {
    void* p = (char*)d_ws + off;
    off += (bytes + 255) & ~(size_t)255;
    return p;
  };
  float*    h      = (float*)alloc((size_t)NN * HH * 4);
  float*    hnew   = (float*)alloc((size_t)NN * HH * 4);
  _Float16* a16    = (_Float16*)alloc((size_t)EE * HH * 2);
  _Float16* hs16   = (_Float16*)alloc((size_t)EE * HH * 2);
  _Float16* w2t    = (_Float16*)alloc((size_t)LL * 65 * 64 * 64 * 2);
  float*    aggr   = (float*)alloc((size_t)NN * HH * 4);
  float*    deg    = (float*)alloc((size_t)NN * 4);
  float*    bns    = (float*)alloc(128 * 4);
  float*    pooled = (float*)alloc((size_t)GG * HH * 4);
  float*    gcnt   = (float*)alloc((size_t)GG * 4);

  // allow >64KB dynamic LDS for the GEMM kernel (WGP has 320KB)
  (void)hipFuncSetAttribute((const void*)k_msg_gemm,
                            hipFuncAttributeMaxDynamicSharedMemorySize,
                            SMEM_BYTES);

  // init counts / pools
  k_zero<<<128, 256, 0, stream>>>(deg, NN);
  k_zero<<<8, 256, 0, stream>>>(gcnt, GG);
  k_zero<<<64, 256, 0, stream>>>(pooled, GG * HH);
  k_deg<<<(EE + 255) / 256, 256, 0, stream>>>(ei, deg);
  k_gcnt<<<(NN + 255) / 256, 256, 0, stream>>>(batch, gcnt);

  // node embedding
  k_node_emb<<<NN / 4, 256, 0, stream>>>(x, nemb_w, nemb_b, h);

  // weight conversion (all layers, incl. bias slice)
  k_convert_w2<<<(LL * 65 * 4096) / 256, 256, 0, stream>>>(ew2, eb2, w2t);

  const int npairs = (NT + 1) / 2;  // 1563 wave pairs of two 16-edge tiles
  const int gblocks = (npairs + WAVES - 1) / WAVES;

  for (int l = 0; l < LL; ++l) {
    k_zero<<<1024, 256, 0, stream>>>(aggr, NN * HH);
    k_zero<<<1, 128, 0, stream>>>(bns, 128);
    k_edge_mlp1<<<EE / 4, 256, 0, stream>>>(
        eattr, ew1 + (size_t)l * EDGE_DIM * HH, eb1 + (size_t)l * HH, h, ei,
        a16, hs16);
    k_msg_gemm<<<gblocks, 256, SMEM_BYTES, stream>>>(
        a16, hs16, w2t + (size_t)l * 65 * 64 * 64, ei, aggr);
    k_root<<<NN / 40, 256, 0, stream>>>(
        aggr, deg, h, rootw + (size_t)l * HH * HH, convb + (size_t)l * HH,
        hnew, bns);
    k_bn_relu<<<NN / 4, 256, 0, stream>>>(
        hnew, bns, bng + (size_t)l * HH, bnb + (size_t)l * HH, h);
  }

  k_pool<<<NN / 4, 256, 0, stream>>>(h, batch, pooled);
  k_readout<<<GG, 64, 0, stream>>>(pooled, gcnt, row1, rob1, row2, rob2, out);
}
